// RelationMemory_21801253995012
// MI455X (gfx1250) — matrix operand order repaired and checked by reading, not compiled, not run on hardware
//
#include <hip/hip_runtime.h>

// ---------------------------------------------------------------------------
// RelationMemory fused kernels for MI455X (gfx1250, wave32, WMMA)
//
// Sizes: B=256, D=128, OUT=1e6, K+1=512, rows R = 512*256 = 131072
// d_out layout (floats): outs_t[R*128] | outs_s[R*128] | mem_v1[OUT*128] | mem_v2[OUT*128]
// ---------------------------------------------------------------------------

#define DIM      128
#define BATCH    256
#define KP1      512
#define ROWS     (KP1 * BATCH)        // 131072
#define OUTROWS  1000000
#define WPITCH   136                   // bf16 elements per LDS W row (272 B, conflict-free)
#define HPITCH   136

typedef __attribute__((ext_vector_type(16))) __bf16 v16bf;
typedef __attribute__((ext_vector_type(8)))  __bf16 v8bf;
typedef __attribute__((ext_vector_type(8)))  float  v8f;
typedef __attribute__((ext_vector_type(4)))  float  v4f;

union BF16x16 { v16bf v; v8bf h[2]; };

__device__ __forceinline__ v8f wmma_bf16(v16bf a, v16bf b, v8f c) {
  // D = A(16x32 bf16) * B(32x16 bf16) + C(16x16 f32)
  return __builtin_amdgcn_wmma_f32_16x16x32_bf16(
      /*neg_a=*/false, a, /*neg_b=*/false, b,
      /*c_mod=*/(short)0, c, /*reuse_a=*/false, /*reuse_b=*/false);
}

// A-fragment (ISA 7.12.2, 16-bit A 16x32): lane(half,m) holds
// K = 8*half + {0..7} (elems 0..7) and K = 16 + 8*half + {0..7} (elems 8..15).
__device__ __forceinline__ v16bf ld_afrag_lds(const __bf16* p) {
  BF16x16 f;
  f.h[0] = *(const v8bf*)(p);        // interleaved low group
  f.h[1] = *(const v8bf*)(p + 16);   // interleaved high group
  return f.v;
}

// B-fragment: lane(half,n) holds K = 8*half + {0..7} (elems 0..7) and
// K = 16 + 8*half + {0..7} (elems 8..15), matching the A-fragment.
__device__ __forceinline__ v16bf ld_bfrag_lds(const __bf16* p) {
  BF16x16 f;
  f.h[0] = *(const v8bf*)(p);
  f.h[1] = *(const v8bf*)(p + 16);
  return f.v;
}

// Build an A-fragment from a global f32 row (gathered memory row), converting
// to bf16: elems 0..7 <- row[lo..lo+7], elems 8..15 <- row[lo+16..lo+23].
__device__ __forceinline__ v16bf cvt_afrag_global(const float* __restrict__ lo) {
  v4f a0 = *(const v4f*)(lo);
  v4f a1 = *(const v4f*)(lo + 4);
  v4f b0 = *(const v4f*)(lo + 16);
  v4f b1 = *(const v4f*)(lo + 20);
  BF16x16 f;
#pragma unroll
  for (int i = 0; i < 4; ++i) {
    f.v[i]      = (__bf16)a0[i];
    f.v[4 + i]  = (__bf16)a1[i];
    f.v[8 + i]  = (__bf16)b0[i];
    f.v[12 + i] = (__bf16)b1[i];
  }
  return f.v;
}

// ---------------------------------------------------------------------------
// 1) Streaming copy of both memory banks into the output (nontemporal).
// ---------------------------------------------------------------------------
__global__ void copy_banks_kernel(const float* __restrict__ m1,
                                  const float* __restrict__ m2,
                                  float* __restrict__ o1,
                                  float* __restrict__ o2) {
  const long n4 = (long)OUTROWS * DIM / 4;  // 32,000,000 float4 per bank
  long i = (long)blockIdx.x * blockDim.x + threadIdx.x;
  const long stride = (long)gridDim.x * blockDim.x;
  const v4f* s1 = (const v4f*)m1;
  const v4f* s2 = (const v4f*)m2;
  v4f* d1 = (v4f*)o1;
  v4f* d2 = (v4f*)o2;
  for (; i < n4; i += stride) {
    __builtin_nontemporal_store(__builtin_nontemporal_load(s1 + i), d1 + i);
    __builtin_nontemporal_store(__builtin_nontemporal_load(s2 + i), d2 + i);
  }
}

// ---------------------------------------------------------------------------
// 2) Momentum update + L2 normalize of the B touched rows (1 wave per row).
//    grid = (256, 2), block = 32.  Runs after copy (stream order).
// ---------------------------------------------------------------------------
__global__ void update_rows_kernel(const float* __restrict__ v1,
                                   const float* __restrict__ v2,
                                   const int* __restrict__ y,
                                   const float* __restrict__ m1,
                                   const float* __restrict__ m2,
                                   float* __restrict__ o1,
                                   float* __restrict__ o2) {
  const int b    = blockIdx.x;
  const int bank = blockIdx.y;
  const int lane = threadIdx.x;          // 0..31, 4 floats each
  const float* v   = bank ? v2 : v1;
  const float* mem = bank ? m2 : m1;
  float*       o   = bank ? o2 : o1;
  const long row = (long)y[b];
  v4f mv = *(const v4f*)(mem + row * DIM + lane * 4);
  v4f vv = *(const v4f*)(v + (long)b * DIM + lane * 4);
  v4f r  = mv * 0.5f + vv * 0.5f;
  float ss = r[0] * r[0] + r[1] * r[1] + r[2] * r[2] + r[3] * r[3];
#pragma unroll
  for (int off = 16; off > 0; off >>= 1) ss += __shfl_xor(ss, off, 32);
  const float inv = 1.0f / sqrtf(ss);
  *(v4f*)(o + row * DIM + lane * 4) = r * inv;
}

// ---------------------------------------------------------------------------
// 3) a = v @ W1^T + b1 for both branches (tiny: 2 x 256x128x128).
//    grid = (256, 2), block = 128.
// ---------------------------------------------------------------------------
__global__ void compute_a_kernel(const float* __restrict__ v1,
                                 const float* __restrict__ v2,
                                 const float* __restrict__ Ws1,
                                 const float* __restrict__ bs1,
                                 const float* __restrict__ Wt1,
                                 const float* __restrict__ bt1,
                                 float* __restrict__ a_s,
                                 float* __restrict__ a_t) {
  const int b  = blockIdx.x;
  const int br = blockIdx.y;               // 0 = s (v1), 1 = t (v2)
  const int e  = threadIdx.x;
  const float* v  = br ? v2 : v1;
  const float* W  = br ? Wt1 : Ws1;
  const float* bb = br ? bt1 : bs1;
  float*       a  = br ? a_t : a_s;
  __shared__ float vrow[DIM];
  vrow[e] = v[b * DIM + e];
  __syncthreads();
  float s = bb[e];
  const float* wr = W + e * DIM;
#pragma unroll 8
  for (int d = 0; d < DIM; ++d) s = fmaf(vrow[d], wr[d], s);
  a[b * DIM + e] = s;
}

// ---------------------------------------------------------------------------
// 4) Fused dual-branch MLP over the gathered rows, WMMA bf16 -> f32.
//    256 threads = 8 waves; each wave owns one 16-row tile of [R,128].
//    grid = R / (16*8) = 1024 blocks.
//    Dynamic LDS: Wbf[4][128][WPITCH] + Hbf[8][16][HPITCH] = 174,080 B.
// ---------------------------------------------------------------------------
__global__ void fused_branch_kernel(const float* __restrict__ mem_v2,
                                    const int* __restrict__ idx,
                                    const float* __restrict__ a_s,
                                    const float* __restrict__ a_t,
                                    const float* __restrict__ Ws2,
                                    const float* __restrict__ bs2,
                                    const float* __restrict__ Wsv,
                                    const float* __restrict__ bsv,
                                    const float* __restrict__ Wt2,
                                    const float* __restrict__ bt2,
                                    const float* __restrict__ Wtv,
                                    const float* __restrict__ btv,
                                    float* __restrict__ out_t,
                                    float* __restrict__ out_s) {
  extern __shared__ char smem[];
  __bf16* Wl = (__bf16*)smem;                                   // 4*128*WPITCH
  __bf16* Hl = (__bf16*)(smem + 4 * 128 * WPITCH * 2);          // 8*16*HPITCH

  const int tid = threadIdx.x;

  // Stage the four 128x128 W matrices into LDS as bf16, padded rows.
#pragma unroll
  for (int mtx = 0; mtx < 4; ++mtx) {
    const float* w = (mtx == 0) ? Wt2 : (mtx == 1) ? Wtv : (mtx == 2) ? Ws2 : Wsv;
    __bf16* dst = Wl + mtx * 128 * WPITCH;
    for (int i = tid; i < 128 * 128; i += 256) {
      const int e = i >> 7, d = i & 127;
      dst[e * WPITCH + d] = (__bf16)w[i];
    }
  }
  __syncthreads();

  const int wave = tid >> 5;
  const int lane = tid & 31;
  const int half = lane >> 4;
  const int ln   = lane & 15;        // A-row m (A/C) and B/C column n

  const int tile = blockIdx.x * 8 + wave;   // 0..8191
  const int r0   = tile * 16;               // first flat row (k*256 + b)
  const int b0   = r0 & 255;                // batch offset of this tile
  const int kk   = r0 >> 8;                 // negatives index (constant in tile)

  __bf16* hl = Hl + wave * 16 * HPITCH;

  // ---- Build gathered-X A-fragments (shared by both branches) -------------
  const int m = ln;                                   // this lane's A row
  const long grow = (long)idx[(long)(b0 + m) * KP1 + kk];
  const float* xrow = mem_v2 + grow * DIM;
  v16bf xf[4];
#pragma unroll
  for (int kc = 0; kc < 4; ++kc)
    xf[kc] = cvt_afrag_global(xrow + kc * 32 + 8 * half);

  // ---- Both branches: 0 = t (v2 / Wt*), 1 = s (v1 / Ws*) ------------------
#pragma unroll 1
  for (int br = 0; br < 2; ++br) {
    const __bf16* W2l = Wl + (br * 2 + 0) * 128 * WPITCH;
    const __bf16* Wvl = Wl + (br * 2 + 1) * 128 * WPITCH;
    const float* aB = br ? a_s : a_t;
    const float* b2 = br ? bs2 : bt2;
    const float* bv = br ? bsv : btv;
    float*       ob = br ? out_s : out_t;

    // Stage 1: T = X @ W2^T ; h = relu(a - T - b2) -> LDS (bf16)
#pragma unroll 1
    for (int nt = 0; nt < 8; ++nt) {
      v8f acc = {};
      const __bf16* wrow = W2l + (nt * 16 + ln) * WPITCH + 8 * half;
#pragma unroll
      for (int kc = 0; kc < 4; ++kc)
        acc = wmma_bf16(xf[kc], ld_bfrag_lds(wrow + kc * 32), acc);
      const float b2v = b2[nt * 16 + ln];
#pragma unroll
      for (int j = 0; j < 8; ++j) {
        const int mm = j + 8 * half;                       // C/D row
        const float av = aB[(b0 + mm) * DIM + nt * 16 + ln];
        float hv = av - acc[j] - b2v;
        hv = hv > 0.0f ? hv : 0.0f;
        hl[mm * HPITCH + nt * 16 + ln] = (__bf16)hv;       // wave-local LDS
      }
    }

    // Rebuild h as A-fragments (same-wave LDS ordering handled by compiler).
    v16bf hf[4];
    const __bf16* hrow = hl + m * HPITCH + 8 * half;
#pragma unroll
    for (int kc = 0; kc < 4; ++kc)
      hf[kc] = ld_afrag_lds(hrow + kc * 32);

    // Stage 2: O = h @ Wv^T + bv -> global
#pragma unroll 1
    for (int nt = 0; nt < 8; ++nt) {
      v8f acc = {};
      const __bf16* wrow = Wvl + (nt * 16 + ln) * WPITCH + 8 * half;
#pragma unroll
      for (int kc = 0; kc < 4; ++kc)
        acc = wmma_bf16(hf[kc], ld_bfrag_lds(wrow + kc * 32), acc);
      const float bvv = bv[nt * 16 + ln];
#pragma unroll
      for (int j = 0; j < 8; ++j) {
        const int mm = j + 8 * half;
        ob[(long)(r0 + mm) * DIM + nt * 16 + ln] = acc[j] + bvv;
      }
    }
  }
}

// ---------------------------------------------------------------------------
extern "C" void kernel_launch(void* const* d_in, const int* in_sizes, int n_in,
                              void* d_out, int out_size, void* d_ws, size_t ws_size,
                              hipStream_t stream) {
  (void)in_sizes; (void)n_in; (void)out_size; (void)ws_size;
  const float* v1   = (const float*)d_in[0];
  const float* v2   = (const float*)d_in[1];
  const int*   y    = (const int*)d_in[2];
  const int*   idx  = (const int*)d_in[3];
  const float* mem1 = (const float*)d_in[4];
  const float* mem2 = (const float*)d_in[5];
  const float* Ws1  = (const float*)d_in[6];
  const float* bs1  = (const float*)d_in[7];
  const float* Ws2  = (const float*)d_in[8];
  const float* bs2  = (const float*)d_in[9];
  const float* Wsv  = (const float*)d_in[10];
  const float* bsv  = (const float*)d_in[11];
  const float* Wt1  = (const float*)d_in[12];
  const float* bt1  = (const float*)d_in[13];
  const float* Wt2  = (const float*)d_in[14];
  const float* bt2  = (const float*)d_in[15];
  const float* Wtv  = (const float*)d_in[16];
  const float* btv  = (const float*)d_in[17];

  float* out   = (float*)d_out;
  float* out_t = out;                                      // [512,256,128]
  float* out_s = out + (long)ROWS * DIM;                   // [512,256,128]
  float* o1    = out + 2L * ROWS * DIM;                    // new_mem_v1
  float* o2    = o1 + (long)OUTROWS * DIM;                 // new_mem_v2

  float* a_s = (float*)d_ws;                               // [256,128]
  float* a_t = a_s + BATCH * DIM;                          // [256,128]

  // 1) bulk bank copy (nontemporal streaming)
  copy_banks_kernel<<<2048, 256, 0, stream>>>(mem1, mem2, o1, o2);
  // 2) momentum + normalize the touched rows (after copy, stream-ordered)
  update_rows_kernel<<<dim3(256, 2), 32, 0, stream>>>(v1, v2, y, mem1, mem2, o1, o2);
  // 3) a = v @ W1^T + b1
  compute_a_kernel<<<dim3(256, 2), 128, 0, stream>>>(v1, v2, Ws1, bs1, Wt1, bt1, a_s, a_t);
  // 4) fused WMMA dual-branch MLP
  const size_t smbytes = (size_t)(4 * 128 * WPITCH + 8 * 16 * HPITCH) * 2; // 174,080 B
  hipFuncSetAttribute((const void*)fused_branch_kernel,
                      hipFuncAttributeMaxDynamicSharedMemorySize, (int)smbytes);
  fused_branch_kernel<<<1024, 256, smbytes, stream>>>(
      mem2, idx, a_s, a_t, Ws2, bs2, Wsv, bsv, Wt2, bt2, Wtv, btv, out_t, out_s);
}
